// MAB_38817914421342
// MI455X (gfx1250) — compile-verified, matrix operand order checked by reading, not run on hardware
//
#include <hip/hip_runtime.h>
#include <hip/hip_bf16.h>

// ---------------------------------------------------------------------------
// MAB block for MI455X (gfx1250): wave32, bf16 WMMA, async global->LDS copies
// B=16, L=1024, D=512, H=8, HEAD=64
// ---------------------------------------------------------------------------

typedef __attribute__((ext_vector_type(16))) __bf16 v16bf;
typedef __attribute__((ext_vector_type(8)))  __bf16 v8bf;
typedef __attribute__((ext_vector_type(8)))  float  v8f;

#define D_MODEL 512
#define SEQ     1024
#define NBATCH  16
#define NHEAD   8
#define HEADD   64

__device__ __forceinline__ v8f wmma_bf16(v16bf a, v16bf b, v8f c) {
  return __builtin_amdgcn_wmma_f32_16x16x32_bf16(
      false, a, false, b, (short)0, c, false, false);
}

// CDNA5 async copy: 16 bytes global -> LDS, tracked by ASYNCcnt.
__device__ __forceinline__ void async_cp_b128(const void* g, void* l) {
  unsigned lds_off = (unsigned)(size_t)l;              // low 32 bits = LDS offset
  unsigned long long ga = (unsigned long long)(size_t)g;
  asm volatile("global_load_async_to_lds_b128 %0, %1, off"
               :: "v"(lds_off), "v"(ga) : "memory");
}
__device__ __forceinline__ void wait_async0() {
  asm volatile("s_wait_asynccnt 0x0" ::: "memory");
}

// A/B fragment from LDS, K-run contiguous per row.
// Element e of lane l -> k = k0 + (e&7) + 8*(l>>4) + 16*(e>>3).
__device__ __forceinline__ v16bf frag_row(const __bf16* lds, int rbase, int k0, int ldk) {
  const int lane = threadIdx.x & 31;
  const int row  = rbase + (lane & 15);
  const int kb   = k0 + 8 * (lane >> 4);
  const __bf16* p = lds + row * ldk + kb;
  v8bf lo = *(const v8bf*)(p);
  v8bf hi = *(const v8bf*)(p + 16);
  v16bf r;
#pragma unroll
  for (int i = 0; i < 8; ++i) { r[i] = lo[i]; r[i + 8] = hi[i]; }
  return r;
}

// B fragment when B[k][n] lives at lds[k*ldk + n].
__device__ __forceinline__ v16bf frag_col(const __bf16* lds, int nbase, int k0, int ldk) {
  const int lane = threadIdx.x & 31;
  const int col  = nbase + (lane & 15);
  const int kb   = k0 + 8 * (lane >> 4);
  v16bf r;
#pragma unroll
  for (int i = 0; i < 8; ++i) {
    r[i]     = lds[(kb + i) * ldk + col];
    r[i + 8] = lds[(kb + 16 + i) * ldk + col];
  }
  return r;
}

// ---------------------------------------------------------------------------
// Kernel 1: Y(bf16)[M,512] = X(f32)[M,512] @ W(f32)[512,512]^T + bias
// BM=128, BN=64, BK=32. grid (M/128, 512/64), block 256 (8 waves);
// each wave: 2x2 block of 16x16 D tiles -> 4 WMMA per K-step.
// ---------------------------------------------------------------------------
__global__ void k_gemm_proj(const float* __restrict__ X,
                            const float* __restrict__ W,
                            const float* __restrict__ bias,
                            __bf16* __restrict__ Y) {
  __shared__ __align__(16) __bf16 As[128 * 40];
  __shared__ __align__(16) __bf16 Bs[64 * 40];

  const int m0 = blockIdx.x * 128;
  const int n0 = blockIdx.y * 64;
  const int tid = threadIdx.x;
  const int wave = tid >> 5, lane = tid & 31;
  const int mta = (wave & 3) * 2;    // row tiles mta, mta+1
  const int ntb = (wave >> 2) * 2;   // col tiles ntb, ntb+1

  v8f c00 = {}, c01 = {}, c10 = {}, c11 = {};

  const int xr = tid >> 1;           // 0..127
  const int xc = (tid & 1) * 16;     // 0,16
  const int wr = tid >> 2;           // 0..63
  const int wc = (tid & 3) * 8;      // 0,8,16,24

  for (int k0 = 0; k0 < D_MODEL; k0 += 32) {
    { // stage X tile (f32 -> bf16), 16 elems/thread
      const float4* src = (const float4*)(X + (size_t)(m0 + xr) * D_MODEL + k0 + xc);
      float4 a = src[0], b = src[1], c = src[2], d = src[3];
      __bf16* dst = As + xr * 40 + xc;
      dst[0]=(__bf16)a.x; dst[1]=(__bf16)a.y; dst[2]=(__bf16)a.z; dst[3]=(__bf16)a.w;
      dst[4]=(__bf16)b.x; dst[5]=(__bf16)b.y; dst[6]=(__bf16)b.z; dst[7]=(__bf16)b.w;
      dst[8]=(__bf16)c.x; dst[9]=(__bf16)c.y; dst[10]=(__bf16)c.z; dst[11]=(__bf16)c.w;
      dst[12]=(__bf16)d.x; dst[13]=(__bf16)d.y; dst[14]=(__bf16)d.z; dst[15]=(__bf16)d.w;
    }
    { // stage W tile (rows = output features), 8 elems/thread
      const float4* src = (const float4*)(W + (size_t)(n0 + wr) * D_MODEL + k0 + wc);
      float4 a = src[0], b = src[1];
      __bf16* dst = Bs + wr * 40 + wc;
      dst[0]=(__bf16)a.x; dst[1]=(__bf16)a.y; dst[2]=(__bf16)a.z; dst[3]=(__bf16)a.w;
      dst[4]=(__bf16)b.x; dst[5]=(__bf16)b.y; dst[6]=(__bf16)b.z; dst[7]=(__bf16)b.w;
    }
    if (k0 + 32 < D_MODEL) {
      __builtin_prefetch(X + (size_t)(m0 + xr) * D_MODEL + k0 + 32 + xc, 0, 1);
      __builtin_prefetch(W + (size_t)(n0 + wr) * D_MODEL + k0 + 32 + wc, 0, 1);
    }
    __syncthreads();
    v16bf a0 = frag_row(As, mta * 16, 0, 40);
    v16bf a1 = frag_row(As, (mta + 1) * 16, 0, 40);
    v16bf b0 = frag_row(Bs, ntb * 16, 0, 40);
    v16bf b1 = frag_row(Bs, (ntb + 1) * 16, 0, 40);
    c00 = wmma_bf16(a0, b0, c00);
    c01 = wmma_bf16(a0, b1, c01);
    c10 = wmma_bf16(a1, b0, c10);
    c11 = wmma_bf16(a1, b1, c11);
    __syncthreads();
  }

#pragma unroll
  for (int r = 0; r < 8; ++r) {
    const int ro   = r + 8 * (lane >> 4);
    const int col0 = n0 + ntb * 16 + (lane & 15);
    const float bb0 = bias[col0], bb1 = bias[col0 + 16];
    size_t i0 = (size_t)(m0 + mta * 16 + ro) * D_MODEL + col0;
    size_t i1 = (size_t)(m0 + (mta + 1) * 16 + ro) * D_MODEL + col0;
    Y[i0]      = (__bf16)(c00[r] + bb0);
    Y[i0 + 16] = (__bf16)(c01[r] + bb1);
    Y[i1]      = (__bf16)(c10[r] + bb0);
    Y[i1 + 16] = (__bf16)(c11[r] + bb1);
  }
}

// ---------------------------------------------------------------------------
// Kernel 2: flash attention per (batch, head, 64-query block) + q residual.
// Double-buffered async K/V staging; S(f32) and P(bf16) share one LDS region.
// grid (SEQ/64, NHEAD, NBATCH), block 256.
// ---------------------------------------------------------------------------
__global__ void k_attn(const __bf16* __restrict__ Q,
                       const __bf16* __restrict__ K,
                       const __bf16* __restrict__ V,
                       float* __restrict__ X0, float scale) {
  __shared__ __align__(16) __bf16 Qs[64 * 72];
  __shared__ __align__(16) __bf16 Ks[2][64 * 72];
  __shared__ __align__(16) __bf16 Vs[2][64 * 72];
  __shared__ __align__(16) float  Sb[64 * 64];   // S view; P(bf16,[64][72]) aliases it
  __shared__ float rowm[64], rowl[64], rowf[64];
  __bf16* Pb = (__bf16*)Sb;

  const int q0 = blockIdx.x * 64;
  const int h  = blockIdx.y;
  const int b  = blockIdx.z;
  const int tid = threadIdx.x;
  const int wave = tid >> 5, lane = tid & 31;
  const int mt  = wave & 3;
  const int ntb = (wave >> 2) * 2;

  const int lr = tid >> 2;          // 0..63
  const int lc = (tid & 3) * 16;    // 0,16,32,48

  { // async stage Q tile and first K/V tiles
    const __bf16* qs = Q + ((size_t)b * SEQ + q0 + lr) * D_MODEL + h * HEADD + lc;
    async_cp_b128(qs,     Qs + lr * 72 + lc);
    async_cp_b128(qs + 8, Qs + lr * 72 + lc + 8);
    const __bf16* ks = K + ((size_t)b * SEQ + lr) * D_MODEL + h * HEADD + lc;
    async_cp_b128(ks,     &Ks[0][lr * 72 + lc]);
    async_cp_b128(ks + 8, &Ks[0][lr * 72 + lc + 8]);
    const __bf16* vs = V + ((size_t)b * SEQ + lr) * D_MODEL + h * HEADD + lc;
    async_cp_b128(vs,     &Vs[0][lr * 72 + lc]);
    async_cp_b128(vs + 8, &Vs[0][lr * 72 + lc + 8]);
  }
  if (tid < 64) { rowm[tid] = -3.0e38f; rowl[tid] = 0.0f; }
  v8f o0 = {}; v8f o1 = {};
  wait_async0();
  __syncthreads();

  int buf = 0;
  for (int j0 = 0; j0 < SEQ; j0 += 64, buf ^= 1) {
    if (j0 + 64 < SEQ) { // prefetch next K/V tiles into the other buffer
      const __bf16* ks = K + ((size_t)b * SEQ + j0 + 64 + lr) * D_MODEL + h * HEADD + lc;
      async_cp_b128(ks,     &Ks[buf ^ 1][lr * 72 + lc]);
      async_cp_b128(ks + 8, &Ks[buf ^ 1][lr * 72 + lc + 8]);
      const __bf16* vs = V + ((size_t)b * SEQ + j0 + 64 + lr) * D_MODEL + h * HEADD + lc;
      async_cp_b128(vs,     &Vs[buf ^ 1][lr * 72 + lc]);
      async_cp_b128(vs + 8, &Vs[buf ^ 1][lr * 72 + lc + 8]);
    }

    // S = Q K^T (two 16x16 tiles per wave, contraction over HEADD=64)
    v8f s0 = {}; v8f s1 = {};
#pragma unroll
    for (int kk = 0; kk < HEADD; kk += 32) {
      v16bf aq  = frag_row(Qs, mt * 16, kk, 72);
      v16bf bk0 = frag_row(&Ks[buf][0], ntb * 16, kk, 72);
      v16bf bk1 = frag_row(&Ks[buf][0], (ntb + 1) * 16, kk, 72);
      s0 = wmma_bf16(aq, bk0, s0);
      s1 = wmma_bf16(aq, bk1, s1);
    }
#pragma unroll
    for (int r = 0; r < 8; ++r) {
      const int row = mt * 16 + r + 8 * (lane >> 4);
      const int col = ntb * 16 + (lane & 15);
      Sb[row * 64 + col]      = s0[r] * scale;
      Sb[row * 64 + col + 16] = s1[r] * scale;
    }
    __syncthreads();

    // online softmax: 4 threads per row, quad reduction via shfl_xor
    {
      const int srow = tid >> 2;
      const int sc   = (tid & 3) * 16;
      float sv[16];
#pragma unroll
      for (int i = 0; i < 16; ++i) sv[i] = Sb[srow * 64 + sc + i];
      float m_t = sv[0];
#pragma unroll
      for (int i = 1; i < 16; ++i) m_t = fmaxf(m_t, sv[i]);
      m_t = fmaxf(m_t, __shfl_xor(m_t, 1, 32));
      m_t = fmaxf(m_t, __shfl_xor(m_t, 2, 32));
      const float m_old = rowm[srow];
      const float m_new = fmaxf(m_old, m_t);
      const float fac   = __expf(m_old - m_new);
      float psum = 0.0f;
#pragma unroll
      for (int i = 0; i < 16; ++i) { sv[i] = __expf(sv[i] - m_new); psum += sv[i]; }
      psum += __shfl_xor(psum, 1, 32);
      psum += __shfl_xor(psum, 2, 32);
      if ((tid & 3) == 0) {
        rowm[srow] = m_new;
        rowl[srow] = rowl[srow] * fac + psum;
        rowf[srow] = fac;
      }
      __syncthreads();                 // S reads done -> safe to overwrite as P
#pragma unroll
      for (int i = 0; i < 16; ++i) Pb[srow * 72 + sc + i] = (__bf16)sv[i];
    }
    __syncthreads();

    // rescale running O, then O += P V
#pragma unroll
    for (int r = 0; r < 8; ++r) {
      const float f = rowf[mt * 16 + r + 8 * (lane >> 4)];
      o0[r] *= f; o1[r] *= f;
    }
#pragma unroll
    for (int kk = 0; kk < 64; kk += 32) {
      v16bf ap  = frag_row(Pb, mt * 16, kk, 72);
      v16bf bv0 = frag_col(&Vs[buf][0], ntb * 16, kk, 72);
      v16bf bv1 = frag_col(&Vs[buf][0], (ntb + 1) * 16, kk, 72);
      o0 = wmma_bf16(ap, bv0, o0);
      o1 = wmma_bf16(ap, bv1, o1);
    }
    if (j0 + 64 < SEQ) wait_async0();  // next tile resident
    __syncthreads();
  }

  // epilogue: normalize, add projected-q residual, store f32
#pragma unroll
  for (int r = 0; r < 8; ++r) {
    const int row = mt * 16 + r + 8 * (lane >> 4);
    const int col = ntb * 16 + (lane & 15);
    const float inv = 1.0f / rowl[row];
    const size_t base = ((size_t)b * SEQ + q0 + row) * D_MODEL + h * HEADD;
    X0[base + col]      = o0[r] * inv + (float)Q[base + col];
    X0[base + col + 16] = o1[r] * inv + (float)Q[base + col + 16];
  }
}

// ---------------------------------------------------------------------------
// Kernel 3: layernorm over 512 features. One block (256 thr) per row.
// ---------------------------------------------------------------------------
__global__ void k_ln(const float* __restrict__ Xin,
                     const float* __restrict__ gamma,
                     const float* __restrict__ beta,
                     __bf16* __restrict__ Yb, float* __restrict__ Yf) {
  const int row = blockIdx.x;
  const int tid = threadIdx.x;
  const float* x = Xin + (size_t)row * D_MODEL;

  float v0 = x[tid], v1 = x[tid + 256];
  float s = v0 + v1, s2 = v0 * v0 + v1 * v1;
#pragma unroll
  for (int m = 16; m >= 1; m >>= 1) {
    s  += __shfl_xor(s,  m, 32);
    s2 += __shfl_xor(s2, m, 32);
  }
  __shared__ float ssum[8], ssum2[8];
  if ((tid & 31) == 0) { ssum[tid >> 5] = s; ssum2[tid >> 5] = s2; }
  __syncthreads();
  if (tid == 0) {
    float a = 0.f, b = 0.f;
#pragma unroll
    for (int i = 0; i < 8; ++i) { a += ssum[i]; b += ssum2[i]; }
    ssum[0] = a; ssum2[0] = b;
  }
  __syncthreads();
  const float mean = ssum[0] * (1.0f / D_MODEL);
  const float var  = ssum2[0] * (1.0f / D_MODEL) - mean * mean;
  const float rstd = rsqrtf(var + 1e-5f);
#pragma unroll
  for (int i = tid; i < D_MODEL; i += 256) {
    const float y = (x[i] - mean) * rstd * gamma[i] + beta[i];
    if (Yf) Yf[(size_t)row * D_MODEL + i] = y;
    if (Yb) Yb[(size_t)row * D_MODEL + i] = (__bf16)y;
  }
}

// ---------------------------------------------------------------------------
// Kernel 4: T(f32) = Xf + relu( Xb @ Wo^T + bo ); Xb bf16 staged via async LDS
// BM=128, BN=64 like k_gemm_proj.
// ---------------------------------------------------------------------------
__global__ void k_gemm_mlp(const __bf16* __restrict__ Xb,
                           const float* __restrict__ W,
                           const float* __restrict__ bias,
                           const float* __restrict__ Xf,
                           float* __restrict__ T) {
  __shared__ __align__(16) __bf16 As[128 * 40];
  __shared__ __align__(16) __bf16 Bs[64 * 40];

  const int m0 = blockIdx.x * 128;
  const int n0 = blockIdx.y * 64;
  const int tid = threadIdx.x;
  const int wave = tid >> 5, lane = tid & 31;
  const int mta = (wave & 3) * 2;
  const int ntb = (wave >> 2) * 2;

  v8f c00 = {}, c01 = {}, c10 = {}, c11 = {};

  const int xr = tid >> 1;           // 0..127
  const int xc = (tid & 1) * 16;     // 0,16
  const int wr = tid >> 2;           // 0..63
  const int wc = (tid & 3) * 8;      // 0,8,16,24

  for (int k0 = 0; k0 < D_MODEL; k0 += 32) {
    { // X tile already bf16: async copy straight into LDS (CDNA5 path)
      const __bf16* src = Xb + (size_t)(m0 + xr) * D_MODEL + k0 + xc;
      __bf16* dst = As + xr * 40 + xc;
      async_cp_b128(src, dst);
      async_cp_b128(src + 8, dst + 8);
    }
    { // W tile f32 -> bf16
      const float4* src = (const float4*)(W + (size_t)(n0 + wr) * D_MODEL + k0 + wc);
      float4 a = src[0], b = src[1];
      __bf16* dst = Bs + wr * 40 + wc;
      dst[0]=(__bf16)a.x; dst[1]=(__bf16)a.y; dst[2]=(__bf16)a.z; dst[3]=(__bf16)a.w;
      dst[4]=(__bf16)b.x; dst[5]=(__bf16)b.y; dst[6]=(__bf16)b.z; dst[7]=(__bf16)b.w;
    }
    if (k0 + 32 < D_MODEL) {
      __builtin_prefetch(W + (size_t)(n0 + wr) * D_MODEL + k0 + 32 + wc, 0, 1);
    }
    wait_async0();
    __syncthreads();
    v16bf a0 = frag_row(As, mta * 16, 0, 40);
    v16bf a1 = frag_row(As, (mta + 1) * 16, 0, 40);
    v16bf b0 = frag_row(Bs, ntb * 16, 0, 40);
    v16bf b1 = frag_row(Bs, (ntb + 1) * 16, 0, 40);
    c00 = wmma_bf16(a0, b0, c00);
    c01 = wmma_bf16(a0, b1, c01);
    c10 = wmma_bf16(a1, b0, c10);
    c11 = wmma_bf16(a1, b1, c11);
    __syncthreads();
  }

#pragma unroll
  for (int r = 0; r < 8; ++r) {
    const int ro   = r + 8 * (lane >> 4);
    const int col0 = n0 + ntb * 16 + (lane & 15);
    const float bb0 = bias[col0], bb1 = bias[col0 + 16];
    size_t i0 = (size_t)(m0 + mta * 16 + ro) * D_MODEL + col0;
    size_t i1 = (size_t)(m0 + (mta + 1) * 16 + ro) * D_MODEL + col0;
    T[i0]      = Xf[i0]      + fmaxf(c00[r] + bb0, 0.0f);
    T[i0 + 16] = Xf[i0 + 16] + fmaxf(c01[r] + bb1, 0.0f);
    T[i1]      = Xf[i1]      + fmaxf(c10[r] + bb0, 0.0f);
    T[i1 + 16] = Xf[i1 + 16] + fmaxf(c11[r] + bb1, 0.0f);
  }
}

// ---------------------------------------------------------------------------
extern "C" void kernel_launch(void* const* d_in, const int* in_sizes, int n_in,
                              void* d_out, int out_size, void* d_ws, size_t ws_size,
                              hipStream_t stream) {
  const float* query     = (const float*)d_in[0];
  const float* key_value = (const float*)d_in[1];
  const float* Wq = (const float*)d_in[2];
  const float* bq = (const float*)d_in[3];
  const float* Wk = (const float*)d_in[4];
  const float* bk = (const float*)d_in[5];
  const float* Wv = (const float*)d_in[6];
  const float* bv = (const float*)d_in[7];
  const float* Wo = (const float*)d_in[8];
  const float* bo = (const float*)d_in[9];
  const float* g0 = (const float*)d_in[10];
  const float* b0 = (const float*)d_in[11];
  const float* g1 = (const float*)d_in[12];
  const float* b1 = (const float*)d_in[13];
  float* out = (float*)d_out;

  // workspace layout (80 MB total, with reuse):
  //  [0,16M)   qb bf16      -> later reused as xb bf16
  //  [16,48M)  kb,vb bf16   -> later reused as xf f32
  //  [48,80M)  x0 f32       -> later reused as t f32
  char* ws = (char*)d_ws;
  const size_t MB16 = 16ull << 20;
  __bf16* qb = (__bf16*)(ws + 0 * MB16);
  __bf16* kb = (__bf16*)(ws + 1 * MB16);
  __bf16* vb = (__bf16*)(ws + 2 * MB16);
  float*  x0 = (float*) (ws + 3 * MB16);
  __bf16* xb = (__bf16*)(ws + 0 * MB16);
  float*  xf = (float*) (ws + 1 * MB16);
  float*  tb = (float*) (ws + 3 * MB16);

  const int M = NBATCH * SEQ;  // 16384
  dim3 blk(256);
  dim3 gGemm(M / 128, D_MODEL / 64);
  dim3 gAttn(SEQ / 64, NHEAD, NBATCH);
  dim3 gLN(M);

  k_gemm_proj<<<gGemm, blk, 0, stream>>>(query,     Wq, bq, qb);
  k_gemm_proj<<<gGemm, blk, 0, stream>>>(key_value, Wk, bk, kb);
  k_gemm_proj<<<gGemm, blk, 0, stream>>>(key_value, Wv, bv, vb);
  k_attn<<<gAttn, blk, 0, stream>>>(qb, kb, vb, x0, 0.04419417382415922f);
  k_ln<<<gLN, blk, 0, stream>>>(x0, g0, b0, xb, xf);
  k_gemm_mlp<<<gGemm, blk, 0, stream>>>(xb, Wo, bo, xf, tb);
  k_ln<<<gLN, blk, 0, stream>>>(tb, g1, b1, (__bf16*)nullptr, out);
}
